// AGRUCell_48584670052559
// MI455X (gfx1250) — matrix operand order repaired and checked by reading, not compiled, not run on hardware
//
#include <hip/hip_runtime.h>

// ---------------------------------------------------------------------------
// AGRU cell for MI455X (gfx1250, wave32, WMMA).
//   z = sigmoid(x W_z^T + b_z + h U_z^T)
//   r = sigmoid(x W_r^T + b_r + h U_r^T)
//   A = W_h - W_h^T - gamma*I
//   h_new = h + eps * z * tanh(x V_h^T + b_h + (r*h) A^T)
// Strategy: bf16 WMMA (v_wmma_f32_16x16x32_bf16), f32 accumulation.
// Phase 1 fuses 5 GEMMs (shared LHS fragments), phase 2 does the dependent
// GEMM with a fused tanh/update epilogue.
// ---------------------------------------------------------------------------

typedef __bf16 bf16;
typedef __attribute__((ext_vector_type(8)))  __bf16 v8bf;
typedef __attribute__((ext_vector_type(16))) __bf16 v16bf;
typedef __attribute__((ext_vector_type(8)))  float  v8f;

#define HID   1024
#define BATCH 16384
#define GAMMA 0.01f

// ---------------- fragment helpers (layouts per CDNA5 ISA) -------

static __device__ __forceinline__ v16bf cat16(v8bf lo, v8bf hi) {
    v16bf r;
#pragma unroll
    for (int i = 0; i < 8; ++i) { r[i] = lo[i]; r[i + 8] = hi[i]; }
    return r;
}

// A-matrix 16x32 bf16 fragment from row-major src [rows x ld], tile (row0,k0).
// lanes 0-15: row=lane,   K chunks {k0..k0+7, k0+16..k0+23}
// lanes16-31: row=lane-16,K chunks {k0+8..k0+15, k0+24..k0+31}
static __device__ __forceinline__ v16bf load_a_frag(const bf16* __restrict__ src,
                                                    int row0, int k0, int lane) {
    const bf16* q = src + (size_t)(row0 + (lane & 15)) * HID
                        + (k0 + ((lane >> 4) << 3));
    v8bf lo = *(const v8bf*)q;
    v8bf hi = *(const v8bf*)(q + 16);
    return cat16(lo, hi);
}

// B-matrix 32x16 bf16 fragment for X @ W^T with row-major W [N x K]:
// B[k][n] = W[n][k]; lane holds column n0+(lane&15), K chunks
// {k0+8h..k0+8h+7, k0+16+8h..k0+16+8h+7} with h = lane>>4 (matches A).
static __device__ __forceinline__ v16bf load_b_frag(const bf16* __restrict__ w,
                                                    int col0, int k0, int lane) {
    const bf16* q = w + (size_t)(col0 + (lane & 15)) * HID
                      + (k0 + ((lane >> 4) << 3));
    v8bf lo = *(const v8bf*)q;
    v8bf hi = *(const v8bf*)(q + 16);
    return cat16(lo, hi);
}

static __device__ __forceinline__ v8f wmma_bf16(v16bf a, v16bf b, v8f c) {
    // (neg_a, A, neg_b, B, c_mod, C, reuse_a, reuse_b)
    return __builtin_amdgcn_wmma_f32_16x16x32_bf16(false, a, false, b,
                                                   (short)0, c, false, false);
}

static __device__ __forceinline__ float fast_sigmoid(float x) {
    return 1.0f / (1.0f + __expf(-x));
}
static __device__ __forceinline__ float fast_tanh(float x) {
    float e = __expf(-2.0f * fabsf(x));
    float t = (1.0f - e) / (1.0f + e);
    return x < 0.0f ? -t : t;
}

// ---------------- stage 0: conversions ---------------------------------------

__global__ void f32_to_bf16_kernel(const float* __restrict__ src,
                                   bf16* __restrict__ dst, int n) {
    int i = (blockIdx.x * blockDim.x + threadIdx.x) * 8;
    if (i + 8 > n) return;
    float4 a = *(const float4*)(src + i);
    float4 b = *(const float4*)(src + i + 4);
    v8bf o;
    o[0] = (bf16)a.x; o[1] = (bf16)a.y; o[2] = (bf16)a.z; o[3] = (bf16)a.w;
    o[4] = (bf16)b.x; o[5] = (bf16)b.y; o[6] = (bf16)b.z; o[7] = (bf16)b.w;
    *(v8bf*)(dst + i) = o;
}

// A = W_h - W_h^T - gamma*I  (bf16 output)
__global__ void build_A_kernel(const float* __restrict__ Wh,
                               bf16* __restrict__ A) {
    int idx = blockIdx.x * blockDim.x + threadIdx.x;
    int i = idx >> 10, j = idx & (HID - 1);
    float v = Wh[(size_t)i * HID + j] - Wh[(size_t)j * HID + i]
            - (i == j ? GAMMA : 0.0f);
    A[idx] = (bf16)v;
}

// ---------------- stage 1: fused z / r / t GEMMs -----------------------------
// Block tile 128(M) x 64(N); 8 waves arranged 4(M) x 2(N); wave tile 32x32.

#define MMA4(ACC, A0, A1, WPTR)                                        \
    do {                                                               \
        v16bf b0 = load_b_frag((WPTR), n0,      k, lane);              \
        v16bf b1 = load_b_frag((WPTR), n0 + 16, k, lane);              \
        ACC[0][0] = wmma_bf16(A0, b0, ACC[0][0]);                      \
        ACC[0][1] = wmma_bf16(A0, b1, ACC[0][1]);                      \
        ACC[1][0] = wmma_bf16(A1, b0, ACC[1][0]);                      \
        ACC[1][1] = wmma_bf16(A1, b1, ACC[1][1]);                      \
    } while (0)

__global__ __launch_bounds__(256)
void agru_phase1_kernel(const bf16* __restrict__ xb, const bf16* __restrict__ hb,
                        const bf16* __restrict__ Wz, const bf16* __restrict__ Uz,
                        const bf16* __restrict__ Wr, const bf16* __restrict__ Ur,
                        const bf16* __restrict__ Vh,
                        const float* __restrict__ bz, const float* __restrict__ br,
                        const float* __restrict__ bh,
                        const float* __restrict__ h_prev,
                        bf16* __restrict__ z_out, bf16* __restrict__ hr_out,
                        float* __restrict__ t_out) {
    const int lane  = threadIdx.x & 31;
    const int wave  = threadIdx.x >> 5;
    const int m0 = blockIdx.y * 128 + (wave & 3) * 32;
    const int n0 = blockIdx.x * 64  + (wave >> 2) * 32;

    v8f accZ[2][2] = {}, accR[2][2] = {}, accT[2][2] = {};

    for (int k = 0; k < HID; k += 32) {
        v16bf xa0 = load_a_frag(xb, m0,      k, lane);
        v16bf xa1 = load_a_frag(xb, m0 + 16, k, lane);
        v16bf ha0 = load_a_frag(hb, m0,      k, lane);
        v16bf ha1 = load_a_frag(hb, m0 + 16, k, lane);
        MMA4(accZ, xa0, xa1, Wz);   // x W_z^T
        MMA4(accZ, ha0, ha1, Uz);   // + h U_z^T
        MMA4(accR, xa0, xa1, Wr);   // x W_r^T
        MMA4(accR, ha0, ha1, Ur);   // + h U_r^T
        MMA4(accT, xa0, xa1, Vh);   // x V_h^T
    }

    // C/D layout: VGPR v -> row 8*(lane>=16)+v, col lane&15 (per 16x16 subtile)
    const int rbase = (lane >> 4) * 8;
    const int cl    = lane & 15;
#pragma unroll
    for (int mi = 0; mi < 2; ++mi)
#pragma unroll
        for (int ni = 0; ni < 2; ++ni) {
            const int col = n0 + ni * 16 + cl;
            const float bzv = bz[col], brv = br[col], bhv = bh[col];
#pragma unroll
            for (int v = 0; v < 8; ++v) {
                const int row = m0 + mi * 16 + rbase + v;
                const size_t idx = (size_t)row * HID + col;
                float zv = fast_sigmoid(accZ[mi][ni][v] + bzv);
                float rv = fast_sigmoid(accR[mi][ni][v] + brv);
                float hv = h_prev[idx];
                z_out[idx]  = (bf16)zv;
                hr_out[idx] = (bf16)(rv * hv);
                t_out[idx]  = accT[mi][ni][v] + bhv;
            }
        }
}

// ---------------- stage 2: (r*h) @ A^T + fused update ------------------------

__global__ __launch_bounds__(256)
void agru_phase2_kernel(const bf16* __restrict__ hrb, const bf16* __restrict__ Ab,
                        const float* __restrict__ t_in, const bf16* __restrict__ z_in,
                        const float* __restrict__ h_prev,
                        const float* __restrict__ eps_p,
                        float* __restrict__ out) {
    const int lane  = threadIdx.x & 31;
    const int wave  = threadIdx.x >> 5;
    const int m0 = blockIdx.y * 128 + (wave & 3) * 32;
    const int n0 = blockIdx.x * 64  + (wave >> 2) * 32;

    v8f acc[2][2] = {};
    for (int k = 0; k < HID; k += 32) {
        v16bf a0 = load_a_frag(hrb, m0,      k, lane);
        v16bf a1 = load_a_frag(hrb, m0 + 16, k, lane);
        MMA4(acc, a0, a1, Ab);
    }

    const float eps = eps_p[0];
    const int rbase = (lane >> 4) * 8;
    const int cl    = lane & 15;
#pragma unroll
    for (int mi = 0; mi < 2; ++mi)
#pragma unroll
        for (int ni = 0; ni < 2; ++ni) {
            const int col = n0 + ni * 16 + cl;
#pragma unroll
            for (int v = 0; v < 8; ++v) {
                const int row = m0 + mi * 16 + rbase + v;
                const size_t idx = (size_t)row * HID + col;
                float dh = fast_tanh(t_in[idx] + acc[mi][ni][v]);
                out[idx] = h_prev[idx] + eps * (float)z_in[idx] * dh;
            }
        }
}

// ---------------- host launcher ----------------------------------------------

extern "C" void kernel_launch(void* const* d_in, const int* in_sizes, int n_in,
                              void* d_out, int out_size, void* d_ws, size_t ws_size,
                              hipStream_t stream) {
    const float* x      = (const float*)d_in[0];
    const float* h_prev = (const float*)d_in[1];
    const float* W_z    = (const float*)d_in[2];
    const float* b_z    = (const float*)d_in[3];
    const float* U_z    = (const float*)d_in[4];
    const float* W_r    = (const float*)d_in[5];
    const float* b_r    = (const float*)d_in[6];
    const float* U_r    = (const float*)d_in[7];
    const float* V_h    = (const float*)d_in[8];
    const float* b_h    = (const float*)d_in[9];
    const float* W_h    = (const float*)d_in[10];
    const float* eps    = (const float*)d_in[11];
    float* out = (float*)d_out;

    char* ws = (char*)d_ws;
    const size_t W_B  = (size_t)HID * HID * sizeof(bf16);       // 2 MB
    const size_t XH_B = (size_t)BATCH * HID * sizeof(bf16);     // 32 MB
    const size_t T_B  = (size_t)BATCH * HID * sizeof(float);    // 64 MB
    bf16*  wWz = (bf16*)(ws);
    bf16*  wUz = (bf16*)(ws + 1 * W_B);
    bf16*  wWr = (bf16*)(ws + 2 * W_B);
    bf16*  wUr = (bf16*)(ws + 3 * W_B);
    bf16*  wVh = (bf16*)(ws + 4 * W_B);
    bf16*  wA  = (bf16*)(ws + 5 * W_B);
    bf16*  xb  = (bf16*)(ws + 6 * W_B);
    bf16*  hb  = (bf16*)(ws + 6 * W_B + XH_B);
    bf16*  zb  = (bf16*)(ws + 6 * W_B + 2 * XH_B);
    float* tb  = (float*)(ws + 6 * W_B + 3 * XH_B);
    bf16*  hrb = (bf16*)(ws + 6 * W_B + 3 * XH_B + T_B);
    // total: 12 MB + 96 MB + 64 MB + 32 MB = 204 MB of d_ws

    const int NW = HID * HID;       // weight elements
    const int NX = BATCH * HID;     // activation elements

    // stage 0: precision conversion + antisymmetric A
    f32_to_bf16_kernel<<<NW / 8 / 256, 256, 0, stream>>>(W_z, wWz, NW);
    f32_to_bf16_kernel<<<NW / 8 / 256, 256, 0, stream>>>(U_z, wUz, NW);
    f32_to_bf16_kernel<<<NW / 8 / 256, 256, 0, stream>>>(W_r, wWr, NW);
    f32_to_bf16_kernel<<<NW / 8 / 256, 256, 0, stream>>>(U_r, wUr, NW);
    f32_to_bf16_kernel<<<NW / 8 / 256, 256, 0, stream>>>(V_h, wVh, NW);
    build_A_kernel<<<NW / 256, 256, 0, stream>>>(W_h, wA);
    f32_to_bf16_kernel<<<NX / 8 / 256, 256, 0, stream>>>(x, xb, NX);
    f32_to_bf16_kernel<<<NX / 8 / 256, 256, 0, stream>>>(h_prev, hb, NX);

    // stage 1: z, r*h, t   (grid: N tiles of 64, M tiles of 128)
    dim3 grid(HID / 64, BATCH / 128);
    agru_phase1_kernel<<<grid, 256, 0, stream>>>(xb, hb, wWz, wUz, wWr, wUr, wVh,
                                                 b_z, b_r, b_h, h_prev,
                                                 zb, hrb, tb);
    // stage 2: dependent GEMM + fused update
    agru_phase2_kernel<<<grid, 256, 0, stream>>>(hrb, wA, tb, zb, h_prev, eps, out);

    (void)in_sizes; (void)n_in; (void)out_size; (void)ws_size;
}
